// MoE_81003083203577
// MI455X (gfx1250) — matrix operand order repaired and checked by reading, not compiled, not run on hardware
//
#include <hip/hip_runtime.h>
#include <hip/hip_bf16.h>
#include <math.h>

// ---------------------------------------------------------------------------
// MoE (top-2 of 8 experts), H=1024, F=2048, T=4096 tokens.
// Routed (gather/grouped-GEMM/scatter) pipeline using CDNA5 WMMA f16 with
// pre-transposed f16 weights and (if available) async global->LDS staging.
// ---------------------------------------------------------------------------

typedef _Float16 v16h __attribute__((ext_vector_type(16)));
typedef _Float16 v8h  __attribute__((ext_vector_type(8)));
typedef _Float16 v4h  __attribute__((ext_vector_type(4)));
typedef float    v8f  __attribute__((ext_vector_type(8)));
typedef float    v4f  __attribute__((ext_vector_type(4)));
typedef int      v2i  __attribute__((ext_vector_type(2)));
typedef int      v4i  __attribute__((ext_vector_type(4)));

#define T_TOK 4096
#define HID   1024
#define FFN_  2048
#define TWOF  4096
#define NEXP  8
#define NROW  (T_TOK * 2)   // 8192 (token, expert) rows total

// ----- workspace layout (bytes) -----
#define OFF_COUNTS   0ul                           // int[8]
#define OFF_OFFSETS  64ul                          // int[8]
#define OFF_CURSORS  128ul                         // int[8]
#define OFF_TOP2E    256ul                         // int[T*2]
#define OFF_TOP2G    (OFF_TOP2E  + 32768ul)        // float[T*2]
#define OFF_ROWTOK   (OFF_TOP2G  + 32768ul)        // int[NROW]
#define OFF_TOKROW   (OFF_ROWTOK + 32768ul)        // int[T*2]
#define OFF_XH       (OFF_TOKROW + 32768ul)        // f16[T*HID]        (8 MB)
#define OFF_ACT      (OFF_XH + 2ul*T_TOK*HID)      // f16[NROW*FFN_]    (33.5 MB)
#define OFF_CONTRIB  (OFF_ACT + 2ul*NROW*FFN_)     // f16[NROW*HID]     (16.8 MB)
#define OFF_WINT     (OFF_CONTRIB + 2ul*NROW*HID)  // f16[E][TWOF][HID] (67 MB, n-major)
#define OFF_WOUTT    (OFF_WINT + 2ul*NEXP*HID*TWOF)// f16[E][HID][FFN_] (33.5 MB, n-major)

// ----- async global->LDS staging (gfx1250) with safe fallback -----
#if defined(__has_builtin)
#if __has_builtin(__builtin_amdgcn_global_load_async_to_lds_b64) && \
    __has_builtin(__builtin_amdgcn_global_load_async_to_lds_b128)
#define HAVE_ASYNC_LDS 1
#endif
#endif

#ifdef HAVE_ASYNC_LDS
typedef __attribute__((address_space(1))) v2i g_v2i;
typedef __attribute__((address_space(3))) v2i l_v2i;
typedef __attribute__((address_space(1))) v4i g_v4i;
typedef __attribute__((address_space(3))) v4i l_v4i;
__device__ __forceinline__ void async_cp_b64(const void* g, void* l) {
  __builtin_amdgcn_global_load_async_to_lds_b64((g_v2i*)g, (l_v2i*)l, 0, 0);
}
__device__ __forceinline__ void async_cp_b128(const void* g, void* l) {
  __builtin_amdgcn_global_load_async_to_lds_b128((g_v4i*)g, (l_v4i*)l, 0, 0);
}
__device__ __forceinline__ void wait_async() {
#if __has_builtin(__builtin_amdgcn_s_wait_asynccnt)
  __builtin_amdgcn_s_wait_asynccnt(0);
#else
  asm volatile("s_wait_asynccnt 0x0" ::: "memory");
#endif
}
#else
__device__ __forceinline__ void async_cp_b64(const void* g, void* l) {
  *(v4h*)l = *(const v4h*)g;
}
__device__ __forceinline__ void async_cp_b128(const void* g, void* l) {
  *(v8h*)l = *(const v8h*)g;
}
__device__ __forceinline__ void wait_async() {}
#endif

__device__ __forceinline__ v16h cat8(v8h lo, v8h hi) {
  return __builtin_shufflevector(lo, hi, 0,1,2,3,4,5,6,7,8,9,10,11,12,13,14,15);
}

// ---------------------------------------------------------------------------
// K0: convert x -> f16, zero counters + loss
// ---------------------------------------------------------------------------
__global__ __launch_bounds__(256) void moe_init(const float* __restrict__ x,
                                                _Float16* __restrict__ xh,
                                                int* counts, int* cursors,
                                                float* loss_out) {
  size_t i = (size_t)blockIdx.x * 256 + threadIdx.x;
  if (i < (size_t)T_TOK * HID) xh[i] = (_Float16)x[i];
  if (i < NEXP) { counts[i] = 0; cursors[i] = 0; }
  if (i == 0) loss_out[0] = 0.0f;
}

// ---------------------------------------------------------------------------
// K0b: fp32 -> f16 transposed weight conversion.  w:[K][N] -> wt:[N][K]
// (per expert slab via blockIdx.z); tiled 32x32 through LDS, coalesced.
// ---------------------------------------------------------------------------
__global__ __launch_bounds__(256) void moe_wconv(const float* __restrict__ w,
                                                 _Float16* __restrict__ wt,
                                                 int K, int N) {
  __shared__ _Float16 tile[32][33];
  const int n0 = blockIdx.x * 32, k0 = blockIdx.y * 32;
  const float* wb = w + (size_t)blockIdx.z * K * N;
  _Float16* wtb = wt + (size_t)blockIdx.z * K * N;
  const int c = threadIdx.x & 31;
  const int r = threadIdx.x >> 5;   // 0..7
#pragma unroll
  for (int i = 0; i < 4; ++i) {
    int kk = r + i * 8;
    tile[kk][c] = (_Float16)wb[(size_t)(k0 + kk) * N + n0 + c];
  }
  __syncthreads();
#pragma unroll
  for (int i = 0; i < 4; ++i) {
    int nn = r + i * 8;
    wtb[(size_t)(n0 + nn) * K + k0 + c] = tile[c][nn];
  }
}

// ---------------------------------------------------------------------------
// K1: router — one wave per token (wave32), top-2 + softmax, count experts
// ---------------------------------------------------------------------------
__global__ __launch_bounds__(256) void moe_router(const float* __restrict__ x,
                                                  const float* __restrict__ w_router,
                                                  int* counts,
                                                  int* __restrict__ top2e,
                                                  float* __restrict__ top2g) {
  __shared__ float s_wr[HID * NEXP];   // 32 KB
  const int tid = threadIdx.x;
  for (int i = tid; i < HID * NEXP; i += 256) s_wr[i] = w_router[i];
  __syncthreads();

  const int wave = tid >> 5, lane = tid & 31;
  const int t = blockIdx.x * 8 + wave;

  float acc[NEXP];
#pragma unroll
  for (int e = 0; e < NEXP; ++e) acc[e] = 0.0f;

  const float* xr = x + (size_t)t * HID;
  for (int k = lane; k < HID; k += 32) {
    float xv = xr[k];
#pragma unroll
    for (int e = 0; e < NEXP; ++e) acc[e] += xv * s_wr[k * NEXP + e];
  }
#pragma unroll
  for (int e = 0; e < NEXP; ++e) {
#pragma unroll
    for (int off = 16; off > 0; off >>= 1)
      acc[e] += __shfl_xor(acc[e], off, 32);
  }

  if (lane == 0) {
    int e0 = 0; float v0 = acc[0];
#pragma unroll
    for (int e = 1; e < NEXP; ++e) if (acc[e] > v0) { v0 = acc[e]; e0 = e; }
    int e1 = (e0 == 0) ? 1 : 0; float v1 = acc[e1];
#pragma unroll
    for (int e = 0; e < NEXP; ++e)
      if (e != e0 && acc[e] > v1) { v1 = acc[e]; e1 = e; }
    float g0 = 1.0f / (1.0f + __expf(v1 - v0));   // softmax over [v0, v1]
    top2e[2 * t]     = e0;  top2e[2 * t + 1] = e1;
    top2g[2 * t]     = g0;  top2g[2 * t + 1] = 1.0f - g0;
    atomicAdd(&counts[e0], 1);
    atomicAdd(&counts[e1], 1);
  }
}

// ---------------------------------------------------------------------------
// K2: exclusive prefix sum over 8 expert counts
// ---------------------------------------------------------------------------
__global__ void moe_scan(const int* __restrict__ counts, int* __restrict__ offsets) {
  if (threadIdx.x == 0 && blockIdx.x == 0) {
    int off = 0;
#pragma unroll
    for (int e = 0; e < NEXP; ++e) { offsets[e] = off; off += counts[e]; }
  }
}

// ---------------------------------------------------------------------------
// K3: fill compacted per-expert row lists + per-token row map
// ---------------------------------------------------------------------------
__global__ __launch_bounds__(256) void moe_fill(const int* __restrict__ top2e,
                                                const int* __restrict__ offsets,
                                                int* cursors,
                                                int* __restrict__ rowTok,
                                                int* __restrict__ tokRow) {
  int t = blockIdx.x * 256 + threadIdx.x;
  if (t >= T_TOK) return;
#pragma unroll
  for (int k = 0; k < 2; ++k) {
    int e = top2e[2 * t + k];
    int pos = offsets[e] + atomicAdd(&cursors[e], 1);
    rowTok[pos] = t;
    tokRow[2 * t + k] = pos;
  }
}

// ---------------------------------------------------------------------------
// K4: GEMM1 + GLU.  act[row, 0:F] = silu(x@Win[:, n]) * (x@Win[:, n+F])
// Block tile: 32 rows x 128 cols (of F); 8 waves (2 M x 4 N); K-step = 32.
// B comes from pre-transposed f16 weights (wint[e][n][k]) -> pure async copy.
// ---------------------------------------------------------------------------
__global__ __launch_bounds__(256) void moe_gemm1(const _Float16* __restrict__ xh,
                                                 const _Float16* __restrict__ wint,
                                                 const int* __restrict__ counts,
                                                 const int* __restrict__ offsets,
                                                 const int* __restrict__ rowTok,
                                                 _Float16* __restrict__ act) {
  const int e = blockIdx.z;
  const int cnt = counts[e];
  const int mtile = blockIdx.y;
  if (mtile * 32 >= cnt) return;
  const int rem = min(32, cnt - mtile * 32);
  const int rowBase = offsets[e] + mtile * 32;
  const int n0 = blockIdx.x * 128;

  __shared__ __align__(16) _Float16 ldsA[32 * 40];
  __shared__ __align__(16) _Float16 ldsB1[128 * 40];
  __shared__ __align__(16) _Float16 ldsB2[128 * 40];
  __shared__ int s_tok[32];

  const int tid = threadIdx.x;
  const int lane = tid & 31;
  const int wave = tid >> 5;
  const int waveM = wave >> 2;   // 0..1
  const int waveN = wave & 3;    // 0..3

  if (tid < 32) s_tok[tid] = rowTok[rowBase + min(tid, rem - 1)];
  __syncthreads();

  const _Float16* wb = wint + (size_t)e * HID * TWOF;   // [n][k], n in [0,4096)

  v8f zero8 = {0.f, 0.f, 0.f, 0.f, 0.f, 0.f, 0.f, 0.f};
  v8f acc1[2] = {zero8, zero8};
  v8f acc2[2] = {zero8, zero8};

  // A-staging assignment: 32 rows x 32 halves, b64 per thread
  const int la_r = tid >> 3;            // 0..31
  const int la_c = (tid & 7) * 4;       // 0,4,..,28 halves
  const _Float16* aSrc = xh + (size_t)s_tok[la_r] * HID + la_c;   // hoisted
  _Float16* aDst = &ldsA[la_r * 40 + la_c];

#pragma unroll 1
  for (int k0 = 0; k0 < HID; k0 += 32) {
    // ---- stage A (gathered token rows, f16, async copy) ----
    async_cp_b64(aSrc + k0, aDst);
    // ---- stage B1/B2 from transposed f16 weights (pure copy) ----
#pragma unroll
    for (int i = 0; i < 2; ++i) {
      int u = tid + i * 256;            // 0..511 units of 8 halves
      int row = u >> 2;                 // 0..127
      int ch = (u & 3) * 8;             // 0,8,16,24 halves
      async_cp_b128(wb + (size_t)(n0 + row) * HID + k0 + ch,
                    &ldsB1[row * 40 + ch]);
      async_cp_b128(wb + (size_t)(FFN_ + n0 + row) * HID + k0 + ch,
                    &ldsB2[row * 40 + ch]);
    }
    if (k0 + 32 < HID) {  // speculative prefetch of next weight panel
      __builtin_prefetch(wb + (size_t)(n0 + (tid >> 1)) * HID + k0 + 32, 0, 1);
    }
    wait_async();
    __syncthreads();

    // ---- fragments + WMMA ----
    const int m = waveM * 16 + (lane & 15);
    const int kA0 = (lane < 16) ? 0 : 16;
    v8h alo = *(const v8h*)&ldsA[m * 40 + kA0];
    v8h ahi = *(const v8h*)&ldsA[m * 40 + kA0 + 8];
    v16h a = cat8(alo, ahi);

    const int kB0 = (lane < 16) ? 0 : 16;
#pragma unroll
    for (int t = 0; t < 2; ++t) {
      int n = waveN * 32 + t * 16 + (lane & 15);
      v8h b1lo = *(const v8h*)&ldsB1[n * 40 + kB0];
      v8h b1hi = *(const v8h*)&ldsB1[n * 40 + kB0 + 8];
      acc1[t] = __builtin_amdgcn_wmma_f32_16x16x32_f16(
          false, a, false, cat8(b1lo, b1hi), (short)0, acc1[t], false, false);
      v8h b2lo = *(const v8h*)&ldsB2[n * 40 + kB0];
      v8h b2hi = *(const v8h*)&ldsB2[n * 40 + kB0 + 8];
      acc2[t] = __builtin_amdgcn_wmma_f32_16x16x32_f16(
          false, a, false, cat8(b2lo, b2hi), (short)0, acc2[t], false, false);
    }
    __syncthreads();
  }

  // ---- epilogue: silu(h1) * g -> act (f16) ----
#pragma unroll
  for (int t = 0; t < 2; ++t) {
#pragma unroll
    for (int i = 0; i < 8; ++i) {
      int mrow = waveM * 16 + ((lane < 16) ? i : i + 8);
      if (mrow < rem) {
        int col = n0 + waveN * 32 + t * 16 + (lane & 15);
        float h1 = acc1[t][i];
        float g  = acc2[t][i];
        float s  = h1 / (1.0f + __expf(-h1));       // silu
        act[(size_t)(rowBase + mrow) * FFN_ + col] = (_Float16)(s * g);
      }
    }
  }
}

// ---------------------------------------------------------------------------
// K5: GEMM2.  contrib[row, 0:H] = act[row, :] @ Wout   (raw, gate applied later)
// B from pre-transposed f16 woutt[e][n][k], n in [0,1024), k in [0,2048).
// ---------------------------------------------------------------------------
__global__ __launch_bounds__(256) void moe_gemm2(const _Float16* __restrict__ act,
                                                 const _Float16* __restrict__ woutt,
                                                 const int* __restrict__ counts,
                                                 const int* __restrict__ offsets,
                                                 _Float16* __restrict__ contrib) {
  const int e = blockIdx.z;
  const int cnt = counts[e];
  const int mtile = blockIdx.y;
  if (mtile * 32 >= cnt) return;
  const int rem = min(32, cnt - mtile * 32);
  const int rowBase = offsets[e] + mtile * 32;
  const int n0 = blockIdx.x * 128;

  __shared__ __align__(16) _Float16 ldsA[32 * 40];
  __shared__ __align__(16) _Float16 ldsB[128 * 40];

  const int tid = threadIdx.x;
  const int lane = tid & 31;
  const int wave = tid >> 5;
  const int waveM = wave >> 2;
  const int waveN = wave & 3;

  const _Float16* wb = woutt + (size_t)e * FFN_ * HID;   // [n][k]

  v8f zero8 = {0.f, 0.f, 0.f, 0.f, 0.f, 0.f, 0.f, 0.f};
  v8f acc[2] = {zero8, zero8};

  const int la_r = tid >> 3;
  const int la_c = (tid & 7) * 4;
  const _Float16* aSrc =
      act + (size_t)(rowBase + min(la_r, rem - 1)) * FFN_ + la_c;
  _Float16* aDst = &ldsA[la_r * 40 + la_c];

#pragma unroll 1
  for (int k0 = 0; k0 < FFN_; k0 += 32) {
    async_cp_b64(aSrc + k0, aDst);
#pragma unroll
    for (int i = 0; i < 2; ++i) {
      int u = tid + i * 256;
      int row = u >> 2;
      int ch = (u & 3) * 8;
      async_cp_b128(wb + (size_t)(n0 + row) * FFN_ + k0 + ch,
                    &ldsB[row * 40 + ch]);
    }
    if (k0 + 32 < FFN_) {
      __builtin_prefetch(wb + (size_t)(n0 + (tid >> 1)) * FFN_ + k0 + 32, 0, 1);
    }
    wait_async();
    __syncthreads();

    const int m = waveM * 16 + (lane & 15);
    const int kA0 = (lane < 16) ? 0 : 16;
    v8h alo = *(const v8h*)&ldsA[m * 40 + kA0];
    v8h ahi = *(const v8h*)&ldsA[m * 40 + kA0 + 8];
    v16h a = cat8(alo, ahi);

    const int kB0 = (lane < 16) ? 0 : 16;
#pragma unroll
    for (int t = 0; t < 2; ++t) {
      int n = waveN * 32 + t * 16 + (lane & 15);
      v8h blo = *(const v8h*)&ldsB[n * 40 + kB0];
      v8h bhi = *(const v8h*)&ldsB[n * 40 + kB0 + 8];
      acc[t] = __builtin_amdgcn_wmma_f32_16x16x32_f16(
          false, a, false, cat8(blo, bhi), (short)0, acc[t], false, false);
    }
    __syncthreads();
  }

#pragma unroll
  for (int t = 0; t < 2; ++t) {
#pragma unroll
    for (int i = 0; i < 8; ++i) {
      int mrow = waveM * 16 + ((lane < 16) ? i : i + 8);
      if (mrow < rem) {
        int col = n0 + waveN * 32 + t * 16 + (lane & 15);
        contrib[(size_t)(rowBase + mrow) * HID + col] = (_Float16)acc[t][i];
      }
    }
  }
}

// ---------------------------------------------------------------------------
// K6: combine — y = bias + g0*contrib[row0] + g1*contrib[row1]  (fixed order)
// ---------------------------------------------------------------------------
__global__ __launch_bounds__(256) void moe_combine(const _Float16* __restrict__ contrib,
                                                   const int* __restrict__ tokRow,
                                                   const float* __restrict__ top2g,
                                                   const float* __restrict__ bias,
                                                   float* __restrict__ y) {
  int i = blockIdx.x * 256 + threadIdx.x;      // float4 groups, total T*H/4
  int t = i >> 8;                               // H/4 = 256 groups per token
  int c = (i & 255);
  int r0 = tokRow[2 * t], r1 = tokRow[2 * t + 1];
  float g0 = top2g[2 * t], g1 = top2g[2 * t + 1];
  v4h a = *(const v4h*)(contrib + (size_t)r0 * HID + c * 4);
  v4h b = *(const v4h*)(contrib + (size_t)r1 * HID + c * 4);
  v4f bv = *(const v4f*)(bias + c * 4);
  v4f out;
#pragma unroll
  for (int j = 0; j < 4; ++j)
    out[j] = bv[j] + g0 * (float)a[j] + g1 * (float)b[j];
  *(v4f*)(y + (size_t)t * HID + c * 4) = out;
}

// ---------------------------------------------------------------------------
extern "C" void kernel_launch(void* const* d_in, const int* in_sizes, int n_in,
                              void* d_out, int out_size, void* d_ws, size_t ws_size,
                              hipStream_t stream) {
  const float* x        = (const float*)d_in[0];
  const float* w_router = (const float*)d_in[1];
  const float* w_in     = (const float*)d_in[2];
  const float* w_out    = (const float*)d_in[3];
  const float* bias     = (const float*)d_in[4];
  float* y = (float*)d_out;

  char* ws = (char*)d_ws;
  int*      counts  = (int*)(ws + OFF_COUNTS);
  int*      offsets = (int*)(ws + OFF_OFFSETS);
  int*      cursors = (int*)(ws + OFF_CURSORS);
  int*      top2e   = (int*)(ws + OFF_TOP2E);
  float*    top2g   = (float*)(ws + OFF_TOP2G);
  int*      rowTok  = (int*)(ws + OFF_ROWTOK);
  int*      tokRow  = (int*)(ws + OFF_TOKROW);
  _Float16* xh      = (_Float16*)(ws + OFF_XH);
  _Float16* act     = (_Float16*)(ws + OFF_ACT);
  _Float16* contrib = (_Float16*)(ws + OFF_CONTRIB);
  _Float16* wint    = (_Float16*)(ws + OFF_WINT);
  _Float16* woutt   = (_Float16*)(ws + OFF_WOUTT);

  float* loss_out = y + (size_t)T_TOK * HID;

  moe_init<<<(T_TOK * HID) / 256, 256, 0, stream>>>(x, xh, counts, cursors, loss_out);
  // w_in:  [H][2F] -> wint:  [2F][H] per expert
  moe_wconv<<<dim3(TWOF / 32, HID / 32, NEXP), 256, 0, stream>>>(w_in, wint, HID, TWOF);
  // w_out: [F][H]  -> woutt: [H][F] per expert
  moe_wconv<<<dim3(HID / 32, FFN_ / 32, NEXP), 256, 0, stream>>>(w_out, woutt, FFN_, HID);
  moe_router<<<T_TOK / 8, 256, 0, stream>>>(x, w_router, counts, top2e, top2g);
  moe_scan<<<1, 32, 0, stream>>>(counts, offsets);
  moe_fill<<<T_TOK / 256, 256, 0, stream>>>(top2e, offsets, cursors, rowTok, tokRow);
  moe_gemm1<<<dim3(FFN_ / 128, T_TOK / 32, NEXP), 256, 0, stream>>>(
      xh, wint, counts, offsets, rowTok, act);
  moe_gemm2<<<dim3(HID / 128, T_TOK / 32, NEXP), 256, 0, stream>>>(
      act, woutt, counts, offsets, contrib);
  moe_combine<<<(T_TOK * HID / 4) / 256, 256, 0, stream>>>(
      contrib, tokRow, top2g, bias, y);
}
